// TelgenTripartiteHeteroGNN_747324309728
// MI455X (gfx1250) — matrix operand order repaired and checked by reading, not compiled, not run on hardware
//
#include <hip/hip_runtime.h>
#include <stdint.h>
#include <stddef.h>

// ---------------------------------------------------------------------------
// Tripartite hetero-GNN forward for MI455X (gfx1250, wave32, WMMA).
// All dense math runs on v_wmma_f32_16x16x32_f16 (f16 A/B, f32 accum).
// Edge aggregation is CSR-gather (no float atomics), fused with the conv MLP.
// Conv kernel uses 32-row M-blocking: each B fragment feeds two WMMAs,
// halving L2 weight traffic versus a 16-row tile.
// ---------------------------------------------------------------------------

#define TWOH 256
#define HIDN 128

typedef __attribute__((ext_vector_type(16))) _Float16 v16h;
typedef __attribute__((ext_vector_type(8)))  _Float16 v8h;
typedef __attribute__((ext_vector_type(8)))  float    v8f;

union Frag16 { v16h v; v8h h[2]; };

#define V8F_ZERO {0.f,0.f,0.f,0.f,0.f,0.f,0.f,0.f}

__device__ __forceinline__ v8f wmma16(v16h a, v16h b, v8f c) {
  // D = A(16x32 f16) * B(32x16 f16) + C(16x16 f32)
  return __builtin_amdgcn_wmma_f32_16x16x32_f16(false, a, false, b, (short)0, c,
                                                false, false);
}

// A fragment (16x32 f16). Lane l: row = l&15, hi = l>>4.
// halves 0..15 -> K = kb + hi*16 + (0..15)
__device__ __forceinline__ v16h ldA(const _Float16* rowBase, int kb, int hi) {
  Frag16 f;
  f.h[0] = *(const v8h*)(rowBase + kb + hi * 16);
  f.h[1] = *(const v8h*)(rowBase + kb + hi * 16 + 8);
  return f.v;
}
// B fragment (32x16 f16) from transposed weights BT[n][K] (contiguous in K).
// Lane l: col n = l&15 (caller bakes into colBase), halves = K = kb + hi*16 + 0..15
__device__ __forceinline__ v16h ldB(const _Float16* colBase, int kb, int hi) {
  Frag16 f;
  const _Float16* p = colBase + kb + hi * 16;
  f.h[0] = *(const v8h*)p;
  f.h[1] = *(const v8h*)(p + 8);
  return f.v;
}

// ------------------------------- utilities ---------------------------------

__global__ void k_zero(uint32_t* __restrict__ p, int n) {
  int i = blockIdx.x * blockDim.x + threadIdx.x;
  if (i < n) p[i] = 0u;
}

// src f32 [K,N] row-major -> dst f16 [N,K] row-major (transposed)
__global__ void k_transpose(const float* __restrict__ src, _Float16* __restrict__ dst,
                            int K, int N) {
  int i = blockIdx.x * blockDim.x + threadIdx.x;
  if (i < K * N) {
    int k = i / N, n = i - k * N;
    dst[(size_t)n * K + k] = (_Float16)src[i];
  }
}

__global__ void k_count(const int* __restrict__ dst, int* __restrict__ cnt, int E) {
  int i = blockIdx.x * blockDim.x + threadIdx.x;
  if (i < E) atomicAdd(&cnt[dst[i]], 1);
}

// Single-block exclusive scan of cnt -> rowptr/cursor; inv = 1/max(cnt,1).
__launch_bounds__(1024)
__global__ void k_scan(const int* __restrict__ cnt, int* __restrict__ rowptr,
                       int* __restrict__ cursor, float* __restrict__ inv, int n) {
  __shared__ int s[1024];
  __shared__ int carry;
  int tid = threadIdx.x;
  if (tid == 0) carry = 0;
  __syncthreads();
  for (int base = 0; base < n; base += 1024) {
    int i = base + tid;
    int v = (i < n) ? cnt[i] : 0;
    s[tid] = v;
    __syncthreads();
    for (int off = 1; off < 1024; off <<= 1) {
      int t = (tid >= off) ? s[tid - off] : 0;
      __syncthreads();
      s[tid] += t;
      __syncthreads();
    }
    int excl = carry + s[tid] - v;
    if (i < n) {
      rowptr[i] = excl;
      cursor[i] = excl;
      inv[i] = 1.0f / fmaxf((float)v, 1.0f);
    }
    __syncthreads();
    if (tid == 0) carry += s[1023];
    __syncthreads();
  }
  if (tid == 0) rowptr[n] = carry;
}

__global__ void k_build(const int* __restrict__ dst, int* __restrict__ cursor,
                        int* __restrict__ eidx, int E) {
  int i = blockIdx.x * blockDim.x + threadIdx.x;
  if (i < E) {
    int slot = atomicAdd(&cursor[dst[i]], 1);
    eidx[slot] = i;
  }
}

// Per-dst-node sum of edge attrs (deterministic CSR order).
__global__ void k_sea(const int* __restrict__ rowptr, const int* __restrict__ eidx,
                      const float* __restrict__ ea, float* __restrict__ sea, int n) {
  int i = blockIdx.x * blockDim.x + threadIdx.x;
  if (i < n) {
    float s = 0.f;
    for (int j = rowptr[i]; j < rowptr[i + 1]; ++j) s += ea[eidx[j]];
    sea[i] = s;
  }
}

__global__ void k_combine(float* __restrict__ h, const float* __restrict__ hn, size_t n) {
  size_t i = (size_t)blockIdx.x * blockDim.x + threadIdx.x;
  if (i < n) h[i] = (fmaxf(hn[i], 0.f) + h[i]) * 0.5f;
}

// ------------------------------- encoder -----------------------------------

// t = x@w1 + b1 ([N,2]@[2,128]); per-block partial sums/sumsq for BN stats.
__global__ void k_enc1(const float* __restrict__ x, const float* __restrict__ w1,
                       const float* __restrict__ b1, float* __restrict__ t,
                       float* __restrict__ partial, int N) {
  __shared__ float red[512];
  int f = threadIdx.x & 127;
  int sub = threadIdx.x >> 7;
  float wa = w1[f], wb = w1[128 + f], bb = b1[f];
  float s = 0.f, s2 = 0.f;
  for (int n = blockIdx.x * 2 + sub; n < N; n += gridDim.x * 2) {
    float v = x[(size_t)n * 2] * wa + x[(size_t)n * 2 + 1] * wb + bb;
    t[(size_t)n * HIDN + f] = v;
    s += v; s2 += v * v;
  }
  red[threadIdx.x] = s;
  red[256 + threadIdx.x] = s2;
  __syncthreads();
  if (sub == 0) {
    partial[(size_t)blockIdx.x * 256 + f]       = red[f] + red[128 + f];
    partial[(size_t)blockIdx.x * 256 + 128 + f] = red[256 + f] + red[256 + 128 + f];
  }
}

// Finalize BN: a = g*rsqrt(var+eps), b = beta - mean*a  (128 threads)
__global__ void k_enc2(const float* __restrict__ partial, int nb,
                       const float* __restrict__ g, const float* __restrict__ beta,
                       float* __restrict__ ab, float Nf) {
  int f = threadIdx.x;
  float s = 0.f, s2 = 0.f;
  for (int b = 0; b < nb; ++b) {
    s  += partial[(size_t)b * 256 + f];
    s2 += partial[(size_t)b * 256 + 128 + f];
  }
  float m = s / Nf;
  float var = s2 / Nf - m * m;
  float a = g[f] * rsqrtf(var + 1e-5f);
  ab[f] = a;
  ab[128 + f] = beta[f] - m * a;
}

// h = relu(bn(t)) @ w2 + b2 : [16,128]@[128,256] per block, WMMA.
__launch_bounds__(256)
__global__ void k_enc3(const float* __restrict__ t, const float* __restrict__ ab,
                       const _Float16* __restrict__ w2T, const float* __restrict__ b2,
                       float* __restrict__ h, int N) {
  __shared__ __align__(16) _Float16 sA[16 * 136];
  __shared__ __align__(16) float sO[16 * 264];
  const int tid = threadIdx.x;
  const int row0 = blockIdx.x * 16;
  for (int it = 0; it < 8; ++it) {
    int idx = it * 256 + tid;
    int r = idx >> 7, c = idx & 127;
    int node = row0 + r;
    float v = 0.f;
    if (node < N) v = fmaxf(t[(size_t)node * HIDN + c] * ab[c] + ab[128 + c], 0.f);
    sA[r * 136 + c] = (_Float16)v;
  }
  __syncthreads();
  const int lane = tid & 31, wv = tid >> 5, r16 = lane & 15, hi = lane >> 4;
  v8f a0 = V8F_ZERO, a1 = V8F_ZERO;
  const _Float16* B0 = w2T + (size_t)(wv * 16 + r16) * 128;
  const _Float16* B1 = w2T + (size_t)(128 + wv * 16 + r16) * 128;
  for (int ks = 0; ks < 4; ++ks) {
    int kb = ks * 32;
    v16h a = ldA(&sA[r16 * 136], kb, hi);
    a0 = wmma16(a, ldB(B0, kb, hi), a0);
    a1 = wmma16(a, ldB(B1, kb, hi), a1);
  }
  int c0 = wv * 16 + r16, c1 = 128 + c0;
  float bb0 = b2[c0], bb1 = b2[c1];
#pragma unroll
  for (int v = 0; v < 8; ++v) {
    sO[(v + 8 * hi) * 264 + c0] = a0[v] + bb0;
    sO[(v + 8 * hi) * 264 + c1] = a1[v] + bb1;
  }
  __syncthreads();
  for (int it = 0; it < 16; ++it) {
    int idx = it * 256 + tid;
    int r = idx >> 8, c = idx & 255;
    int node = row0 + r;
    if (node < N) h[(size_t)node * TWOH + c] = sO[r * 264 + c];
  }
}

// ------------------------- fused conv (gather + MLP) ------------------------
// agg[n] = (sum_e x_src[src[e]] + sea[n]*we + deg*be) / max(deg,1)
// out    = relu(agg@w1+b1)@w2 + b2  -> written into out[:, col_off:col_off+128]
// 32 destination rows per block; each B fragment is reused for two M-tiles.
__launch_bounds__(256)
__global__ void k_conv(const float* __restrict__ xsrc, const int* __restrict__ src,
                       const int* __restrict__ rowptr, const int* __restrict__ eidx,
                       const float* __restrict__ invc, const float* __restrict__ sea,
                       const float* __restrict__ we, const float* __restrict__ be,
                       const _Float16* __restrict__ w1T, const float* __restrict__ b1,
                       const _Float16* __restrict__ w2T, const float* __restrict__ b2,
                       float* __restrict__ out, int col_off, int n_dst) {
  __shared__ __align__(16) _Float16 sA[32 * 264];  // 32 x 256 agg tile (f16)
  __shared__ __align__(16) _Float16 sH[32 * 136];  // 32 x 128 hidden (f16)
  __shared__ __align__(16) float    sO[32 * 136];  // 32 x 128 out (f32)
  __shared__ int sSrc[256];

  const int tid = threadIdx.x;
  const int row0 = blockIdx.x * 32;
  const int f = tid;  // feature 0..255
  const float wef = we[f];
  const float bef = be[f];

  // Stage A: aggregate 32 destination nodes (coalesced row gathers via CSR)
  for (int i = 0; i < 32; ++i) {
    int node = row0 + i;   // uniform across block
    float val = 0.f;
    if (node < n_dst) {
      int beg = rowptr[node], end = rowptr[node + 1];
      float acc = 0.f, accb = 0.f;
      for (int cb = beg; cb < end; cb += 256) {
        int m = end - cb; if (m > 256) m = 256;
        if (tid < m) sSrc[tid] = src[eidx[cb + tid]];
        __syncthreads();
        int u = 0;
        for (; u + 1 < m; u += 2) {
          acc  += xsrc[(size_t)sSrc[u] * TWOH + f];
          accb += xsrc[(size_t)sSrc[u + 1] * TWOH + f];
        }
        if (u < m) acc += xsrc[(size_t)sSrc[u] * TWOH + f];
        __syncthreads();
      }
      float deg = (float)(end - beg);
      val = (acc + accb + sea[node] * wef + deg * bef) * invc[node];
    }
    sA[i * 264 + f] = (_Float16)val;
  }
  __syncthreads();

  const int lane = tid & 31, wv = tid >> 5, r16 = lane & 15, hi = lane >> 4;
  const int col = wv * 16 + r16;

  // Stage B: h1 = relu(A[32,256] @ w1[256,128] + b1), 8 K-steps, 2 M-tiles
  {
    v8f acc0 = V8F_ZERO, acc1 = V8F_ZERO;
    const _Float16* Bb = w1T + (size_t)col * 256;
    for (int ks = 0; ks < 8; ++ks) {
      int kb = ks * 32;
      v16h b = ldB(Bb, kb, hi);
      acc0 = wmma16(ldA(&sA[r16 * 264], kb, hi), b, acc0);
      acc1 = wmma16(ldA(&sA[(16 + r16) * 264], kb, hi), b, acc1);
    }
    float bb = b1[col];
#pragma unroll
    for (int v = 0; v < 8; ++v) {
      int r = v + 8 * hi;
      sH[r * 136 + col]        = (_Float16)fmaxf(acc0[v] + bb, 0.f);
      sH[(16 + r) * 136 + col] = (_Float16)fmaxf(acc1[v] + bb, 0.f);
    }
  }
  __syncthreads();

  // Stage C: out = h1[32,128] @ w2[128,128] + b2, 4 K-steps, 2 M-tiles
  {
    v8f acc0 = V8F_ZERO, acc1 = V8F_ZERO;
    const _Float16* Bb = w2T + (size_t)col * 128;
    for (int ks = 0; ks < 4; ++ks) {
      int kb = ks * 32;
      v16h b = ldB(Bb, kb, hi);
      acc0 = wmma16(ldA(&sH[r16 * 136], kb, hi), b, acc0);
      acc1 = wmma16(ldA(&sH[(16 + r16) * 136], kb, hi), b, acc1);
    }
    float bb = b2[col];
#pragma unroll
    for (int v = 0; v < 8; ++v) {
      int r = v + 8 * hi;
      sO[r * 136 + col]        = acc0[v] + bb;
      sO[(16 + r) * 136 + col] = acc1[v] + bb;
    }
  }
  __syncthreads();

  // coalesced store (512B rows): 32 rows x 128 cols
  for (int it = 0; it < 16; ++it) {
    int idx = it * 256 + tid;
    int r = idx >> 7, c = idx & 127;
    int node = row0 + r;
    if (node < n_dst) out[(size_t)node * TWOH + col_off + c] = sO[r * 136 + c];
  }
}

// ------------------------------ prediction ---------------------------------
// out[n,k] = (relu?)( relu(h[n,:256]@w1+b1) . w2 + b2 )
__launch_bounds__(256)
__global__ void k_pred(const float* __restrict__ h, const _Float16* __restrict__ w1T,
                       const float* __restrict__ b1, const float* __restrict__ w2,
                       const float* __restrict__ b2, float* __restrict__ outp,
                       int kidx, int N, int do_relu) {
  __shared__ __align__(16) _Float16 sA[16 * 264];
  __shared__ __align__(16) _Float16 sH[16 * 136];
  __shared__ float sR[256];
  const int tid = threadIdx.x;
  const int row0 = blockIdx.x * 16;
  for (int it = 0; it < 16; ++it) {
    int idx = it * 256 + tid;
    int r = idx >> 8, c = idx & 255;
    int node = row0 + r;
    float v = (node < N) ? h[(size_t)node * TWOH + c] : 0.f;
    sA[r * 264 + c] = (_Float16)v;
  }
  __syncthreads();
  const int lane = tid & 31, wv = tid >> 5, r16 = lane & 15, hi = lane >> 4;
  v8f acc1 = V8F_ZERO;
  {
    const _Float16* Bb = w1T + (size_t)(wv * 16 + r16) * 256;
    for (int ks = 0; ks < 8; ++ks) {
      int kb = ks * 32;
      acc1 = wmma16(ldA(&sA[r16 * 264], kb, hi), ldB(Bb, kb, hi), acc1);
    }
    int col = wv * 16 + r16;
    float bb = b1[col];
#pragma unroll
    for (int v = 0; v < 8; ++v) {
      float x = fmaxf(acc1[v] + bb, 0.f);
      sH[(v + 8 * hi) * 136 + col] = (_Float16)x;
    }
  }
  __syncthreads();
  int r = tid >> 4, seg = tid & 15;
  float p = 0.f;
#pragma unroll
  for (int i = 0; i < 8; ++i)
    p += (float)sH[r * 136 + seg * 8 + i] * w2[seg * 8 + i];
  sR[tid] = p;
  __syncthreads();
  if (tid < 16) {
    float s = 0.f;
    for (int i = 0; i < 16; ++i) s += sR[tid * 16 + i];
    s += b2[0];
    if (do_relu) s = fmaxf(s, 0.f);
    int node = row0 + tid;
    if (node < N) outp[(size_t)node * 4 + kidx] = s;
  }
}

// ------------------------------- host side ---------------------------------

extern "C" void kernel_launch(void* const* d_in, const int* in_sizes, int n_in,
                              void* d_out, int out_size, void* d_ws, size_t ws_size,
                              hipStream_t stream) {
  (void)n_in; (void)out_size; (void)ws_size;
  auto cdiv = [](int a, int b) { return (a + b - 1) / b; };

  const float* x_cons = (const float*)d_in[0];
  const float* x_vals = (const float*)d_in[1];
  const float* x_obj  = (const float*)d_in[2];
  const int N_CONS = in_sizes[0] / 2;
  const int N_VALS = in_sizes[1] / 2;
  const int N_OBJ  = in_sizes[2] / 2;

  const int*   cv_src = (const int*)d_in[3];  const int* cv_dst = (const int*)d_in[4];
  const float* ea_cv  = (const float*)d_in[5];
  const int*   vc_src = (const int*)d_in[6];  const int* vc_dst = (const int*)d_in[7];
  const float* ea_vc  = (const float*)d_in[8];
  const int*   vo_src = (const int*)d_in[9];  const int* vo_dst = (const int*)d_in[10];
  const float* ea_vo  = (const float*)d_in[11];
  const int*   ov_src = (const int*)d_in[12]; const int* ov_dst = (const int*)d_in[13];
  const float* ea_ov  = (const float*)d_in[14];
  const int*   co_src = (const int*)d_in[15]; const int* co_dst = (const int*)d_in[16];
  const float* ea_co  = (const float*)d_in[17];
  const int*   oc_src = (const int*)d_in[18]; const int* oc_dst = (const int*)d_in[19];
  const float* ea_oc  = (const float*)d_in[20];

  // params flattened jax-pytree style (dict keys sorted): convs, enc, pred_cons, pred_vals
  const int PBASE = 21;                 // convs[j=0..1][et sorted: c2o,c2v,o2c,o2v,v2c,v2o][b1,b2,be,w1,w2,we]
  auto convP = [&](int j, int et, int w) { return (const float*)d_in[PBASE + j * 36 + et * 6 + w]; };
  const int ENCB = PBASE + 72;          // enc: cons,obj,vals  x  [b1,b2,beta,g,w1,w2]
  auto encP = [&](int t, int w) { return (const float*)d_in[ENCB + t * 6 + w]; };
  const int PREDC = ENCB + 18;          // pred_cons: b1,b2,w1,w2
  const int PREDV = PREDC + 4;          // pred_vals: b1,b2,w1,w2

  // -------- workspace arena --------
  char* wsbase = (char*)d_ws;
  size_t offw = 0;
  auto alloc = [&](size_t bytes) -> void* {
    offw = (offw + 255) & ~(size_t)255;
    void* p = wsbase + offw;
    offw += bytes;
    return p;
  };

  float* h_cons  = (float*)alloc((size_t)N_CONS * TWOH * 4);
  float* h_vals  = (float*)alloc((size_t)N_VALS * TWOH * 4);
  float* h_obj   = (float*)alloc((size_t)N_OBJ  * TWOH * 4);
  float* hn_cons = (float*)alloc((size_t)N_CONS * TWOH * 4);
  float* hn_vals = (float*)alloc((size_t)N_VALS * TWOH * 4);
  float* hn_obj  = (float*)alloc((size_t)N_OBJ  * TWOH * 4);
  // encoder temporaries alias the (not-yet-used) hn buffers
  float* t_buf   = hn_vals;       // [maxN,128] f32
  float* partial = hn_cons;       // [160,256] f32
  float* ab      = hn_obj;        // [256] f32

  struct EdgeT {
    const int* src; const int* dst; const float* ea; int E; int n_dst;
    int* cnt; int* rowptr; int* cursor; int* eidx; float* inv; float* sea;
  };
  EdgeT ed[6] = {
    { cv_src, cv_dst, ea_cv, in_sizes[3],  N_VALS, 0,0,0,0,0,0 },  // 0 c2v
    { vc_src, vc_dst, ea_vc, in_sizes[6],  N_CONS, 0,0,0,0,0,0 },  // 1 v2c
    { vo_src, vo_dst, ea_vo, in_sizes[9],  N_OBJ,  0,0,0,0,0,0 },  // 2 v2o
    { ov_src, ov_dst, ea_ov, in_sizes[12], N_VALS, 0,0,0,0,0,0 },  // 3 o2v
    { co_src, co_dst, ea_co, in_sizes[15], N_OBJ,  0,0,0,0,0,0 },  // 4 c2o
    { oc_src, oc_dst, ea_oc, in_sizes[18], N_CONS, 0,0,0,0,0,0 },  // 5 o2c
  };
  for (int i = 0; i < 6; ++i) {
    ed[i].cnt    = (int*)alloc((size_t)ed[i].n_dst * 4);
    ed[i].rowptr = (int*)alloc(((size_t)ed[i].n_dst + 1) * 4);
    ed[i].cursor = (int*)alloc((size_t)ed[i].n_dst * 4);
    ed[i].eidx   = (int*)alloc((size_t)ed[i].E * 4);
    ed[i].inv    = (float*)alloc((size_t)ed[i].n_dst * 4);
    ed[i].sea    = (float*)alloc((size_t)ed[i].n_dst * 4);
  }

  _Float16* cw1T[2][6]; _Float16* cw2T[2][6]; _Float16* ew2T[3]; _Float16* pw1T[2];
  for (int j = 0; j < 2; ++j)
    for (int e = 0; e < 6; ++e) {
      cw1T[j][e] = (_Float16*)alloc((size_t)256 * 128 * 2);
      cw2T[j][e] = (_Float16*)alloc((size_t)128 * 128 * 2);
    }
  for (int t = 0; t < 3; ++t) ew2T[t] = (_Float16*)alloc((size_t)128 * 256 * 2);
  pw1T[0] = (_Float16*)alloc((size_t)256 * 128 * 2);
  pw1T[1] = (_Float16*)alloc((size_t)256 * 128 * 2);

  // -------- weight prep (f32 -> transposed f16) --------
  auto tr = [&](const float* s, _Float16* d, int K, int N) {
    k_transpose<<<cdiv(K * N, 256), 256, 0, stream>>>(s, d, K, N);
  };
  for (int j = 0; j < 2; ++j)
    for (int e = 0; e < 6; ++e) {
      tr(convP(j, e, 3), cw1T[j][e], 256, 128);   // w1 [256,128]
      tr(convP(j, e, 4), cw2T[j][e], 128, 128);   // w2 [128,128]
    }
  for (int t = 0; t < 3; ++t) tr(encP(t, 5), ew2T[t], 128, 256);  // enc w2 [128,256]
  tr((const float*)d_in[PREDC + 2], pw1T[0], 256, 128);
  tr((const float*)d_in[PREDV + 2], pw1T[1], 256, 128);

  // -------- CSR build per edge type --------
  for (int i = 0; i < 6; ++i) {
    EdgeT& e = ed[i];
    k_zero<<<cdiv(e.n_dst, 256), 256, 0, stream>>>((uint32_t*)e.cnt, e.n_dst);
    k_count<<<cdiv(e.E, 256), 256, 0, stream>>>(e.dst, e.cnt, e.E);
    k_scan<<<1, 1024, 0, stream>>>(e.cnt, e.rowptr, e.cursor, e.inv, e.n_dst);
    k_build<<<cdiv(e.E, 256), 256, 0, stream>>>(e.dst, e.cursor, e.eidx, e.E);
    k_sea<<<cdiv(e.n_dst, 256), 256, 0, stream>>>(e.rowptr, e.eidx, e.ea, e.sea, e.n_dst);
  }

  // -------- encoders --------
  auto run_enc = [&](const float* x, int N, int encIdx, float* hout) {
    k_enc1<<<160, 256, 0, stream>>>(x, encP(encIdx, 4), encP(encIdx, 0), t_buf, partial, N);
    k_enc2<<<1, 128, 0, stream>>>(partial, 160, encP(encIdx, 3), encP(encIdx, 2), ab, (float)N);
    k_enc3<<<cdiv(N, 16), 256, 0, stream>>>(t_buf, ab, ew2T[encIdx], encP(encIdx, 1), hout, N);
  };
  run_enc(x_cons, N_CONS, 0, h_cons);
  run_enc(x_obj,  N_OBJ,  1, h_obj);
  run_enc(x_vals, N_VALS, 2, h_vals);

  // -------- message passing + heads --------
  auto run_conv = [&](int j, int pet, EdgeT& e, const float* xsrc, float* outbuf, int col) {
    k_conv<<<cdiv(e.n_dst, 32), 256, 0, stream>>>(
        xsrc, e.src, e.rowptr, e.eidx, e.inv, e.sea,
        convP(j, pet, 5), convP(j, pet, 2),          // we, be
        cw1T[j][pet], convP(j, pet, 0),              // w1T, b1
        cw2T[j][pet], convP(j, pet, 1),              // w2T, b2
        outbuf, col, e.n_dst);
  };

  float* out_vals = (float*)d_out;
  float* out_cons = (float*)d_out + (size_t)N_VALS * 4;

  for (int ko = 0; ko < 4; ++ko) {
    for (int j = 0; j < 2; ++j) {
      // param-et sorted order: 0=c2o 1=c2v 2=o2c 3=o2v 4=v2c 5=v2o
      run_conv(j, 1, ed[0], h_cons, hn_vals, 0);     // c2v
      run_conv(j, 3, ed[3], h_obj,  hn_vals, 128);   // o2v
      run_conv(j, 4, ed[1], h_vals, hn_cons, 0);     // v2c
      run_conv(j, 2, ed[5], h_obj,  hn_cons, 128);   // o2c
      run_conv(j, 5, ed[2], h_vals, hn_obj,  0);     // v2o
      run_conv(j, 0, ed[4], h_cons, hn_obj,  128);   // c2o
      k_combine<<<cdiv(N_VALS * TWOH, 256), 256, 0, stream>>>(h_vals, hn_vals, (size_t)N_VALS * TWOH);
      k_combine<<<cdiv(N_CONS * TWOH, 256), 256, 0, stream>>>(h_cons, hn_cons, (size_t)N_CONS * TWOH);
      k_combine<<<cdiv(N_OBJ  * TWOH, 256), 256, 0, stream>>>(h_obj,  hn_obj,  (size_t)N_OBJ  * TWOH);
    }
    k_pred<<<cdiv(N_VALS, 16), 256, 0, stream>>>(
        h_vals, pw1T[1], (const float*)d_in[PREDV + 0], (const float*)d_in[PREDV + 3],
        (const float*)d_in[PREDV + 1], out_vals, ko, N_VALS, 1);
    k_pred<<<cdiv(N_CONS, 16), 256, 0, stream>>>(
        h_cons, pw1T[0], (const float*)d_in[PREDC + 0], (const float*)d_in[PREDC + 3],
        (const float*)d_in[PREDC + 1], out_cons, ko, N_CONS, 0);
  }
}
